// FaultyLSTM_66915590471997
// MI455X (gfx1250) — matrix operand order repaired and checked by reading, not compiled, not run on hardware
//
#include <hip/hip_runtime.h>

// Problem constants (from reference): B=64, S=512, IN=512, H=1024, L=2
#define LB   64
#define LS   512
#define LIN  512
#define LH   1024
#define LG   (4 * LH)   // 4096 gate rows

typedef __attribute__((ext_vector_type(16))) __bf16 v16bf;
typedef __attribute__((ext_vector_type(8)))  float  v8f;

// ---------------------------------------------------------------------------
// helpers
// ---------------------------------------------------------------------------
static __device__ __forceinline__ v16bf load_bf16x16(const __bf16* p0,
                                                     const __bf16* p1) {
  union { v16bf v; uint4 u[2]; } t;
  t.u[0] = *reinterpret_cast<const uint4*>(p0);
  t.u[1] = *reinterpret_cast<const uint4*>(p1);
  return t.v;
}

static __device__ __forceinline__ float fast_sigmoid(float x) {
  return 1.0f / (1.0f + __expf(-x));
}

// ---------------------------------------------------------------------------
// one-time prep kernels
// ---------------------------------------------------------------------------
__global__ void cvt_f32_to_bf16(const float* __restrict__ src,
                                __bf16* __restrict__ dst, long n) {
  long i = (long)blockIdx.x * blockDim.x + threadIdx.x;
  long stride = (long)gridDim.x * blockDim.x;
  for (; i < n; i += stride) dst[i] = (__bf16)src[i];
}

__global__ void bias_sum_kernel(const float* __restrict__ bi,
                                const float* __restrict__ bh,
                                float* __restrict__ bs, int n) {
  int i = blockIdx.x * blockDim.x + threadIdx.x;
  if (i < n) bs[i] = bi[i] + bh[i];
}

__global__ void zero_f32(float* __restrict__ p, long n) {
  long i = (long)blockIdx.x * blockDim.x + threadIdx.x;
  long stride = (long)gridDim.x * blockDim.x;
  for (; i < n; i += stride) p[i] = 0.0f;
}

__global__ void zero_bf16(__bf16* __restrict__ p, long n) {
  long i = (long)blockIdx.x * blockDim.x + threadIdx.x;
  long stride = (long)gridDim.x * blockDim.x;
  for (; i < n; i += stride) p[i] = (__bf16)0.0f;
}

__global__ void zero_u32(unsigned* __restrict__ p, int n) {
  int i = blockIdx.x * blockDim.x + threadIdx.x;
  if (i < n) p[i] = 0u;
}

// ---------------------------------------------------------------------------
// Device-wide barrier: monotonically increasing generation counter.
// All 64 blocks are co-resident (64 x 128 threads), so spinning is safe.
// ---------------------------------------------------------------------------
static __device__ __forceinline__ void grid_sync(unsigned* bar, unsigned nblk,
                                                 unsigned& gen) {
  __syncthreads();
  gen += 1;
  if (threadIdx.x == 0) {
    __threadfence();  // make this block's h/c stores device-visible
    __hip_atomic_fetch_add(bar, 1u, __ATOMIC_ACQ_REL, __HIP_MEMORY_SCOPE_AGENT);
    while (__hip_atomic_load(bar, __ATOMIC_ACQUIRE, __HIP_MEMORY_SCOPE_AGENT) <
           gen * nblk) {
      __builtin_amdgcn_s_sleep(1);
    }
  }
  __syncthreads();
}

// ---------------------------------------------------------------------------
// One LSTM cell tile: this wave owns 16(batch) x 16(hidden) and keeps FOUR
// f32 WMMA accumulators (i,f,g,o).  gates = A1*W1^T + A2*W2^T + bias.
//   A1: [64 x K1] bf16, row stride lda1   (x_t for layer0, h0_new for layer1)
//   A2: [64 x K2] bf16, row stride LH     (recurrent h of this layer)
//   W1/W2: [4H x K] bf16 row-major; B-column n == W-row (g*H+n): contiguous.
// The W1/W2 tile addresses are t-invariant, letting the compiler keep weight
// tiles resident in the extended 1024-VGPR file across the whole scan.
// ---------------------------------------------------------------------------
static __device__ __forceinline__ void cell_tile(
    const __bf16* A1, long lda1, int K1, const __bf16* W1,
    const __bf16* A2, int K2, const __bf16* W2,
    const float* bsum, float* c, __bf16* h_out, float* hf,
    float* out, long out_stride,
    int m0, int n0, int half, int lc) {
  v8f acc[4];
  #pragma unroll
  for (int g = 0; g < 4; ++g) acc[g] = (v8f){0.f,0.f,0.f,0.f,0.f,0.f,0.f,0.f};

  // ---- GEMM 1: input contribution -------------------------------------
  for (int k0 = 0; k0 < K1; k0 += 32) {
    const __bf16* ap = A1 + (long)(m0 + lc) * lda1 + k0 + half * 8;
    v16bf a = load_bf16x16(ap, ap + 16);
    #pragma unroll
    for (int g = 0; g < 4; ++g) {
      const __bf16* bp = W1 + (long)(g * LH + n0 + lc) * K1 + k0 + half * 8;
      v16bf b = load_bf16x16(bp, bp + 16);
      acc[g] = __builtin_amdgcn_wmma_f32_16x16x32_bf16(
          false, a, false, b, (short)0, acc[g], false, false);
    }
  }

  // ---- GEMM 2: recurrent contribution ---------------------------------
  for (int k0 = 0; k0 < K2; k0 += 32) {
    const __bf16* ap = A2 + (long)(m0 + lc) * LH + k0 + half * 8;
    v16bf a = load_bf16x16(ap, ap + 16);
    #pragma unroll
    for (int g = 0; g < 4; ++g) {
      const __bf16* bp = W2 + (long)(g * LH + n0 + lc) * K2 + k0 + half * 8;
      v16bf b = load_bf16x16(bp, bp + 16);
      acc[g] = __builtin_amdgcn_wmma_f32_16x16x32_bf16(
          false, a, false, b, (short)0, acc[g], false, false);
    }
  }

  // ---- pointwise LSTM cell update -------------------------------------
  // C/D layout: VGPR r -> M = r (lanes 0-15) or 8+r (lanes 16-31); N = lane&15
  const int n = n0 + lc;
  const float bi = bsum[n];
  const float bf = bsum[LH + n];
  const float bg = bsum[2 * LH + n];
  const float bo = bsum[3 * LH + n];
  #pragma unroll
  for (int r = 0; r < 8; ++r) {
    const int  m   = m0 + r + half * 8;
    const long idx = (long)m * LH + n;
    float si = fast_sigmoid(acc[0][r] + bi);
    float sf = fast_sigmoid(acc[1][r] + bf);
    float tg = tanhf(acc[2][r] + bg);
    float so = fast_sigmoid(acc[3][r] + bo);
    float cn = sf * c[idx] + si * tg;
    c[idx]   = cn;
    float hn = so * tanhf(cn);
    h_out[idx] = (__bf16)hn;
    hf[idx]    = hn;
    if (out) out[(long)m * out_stride + n] = hn;
  }
}

// ---------------------------------------------------------------------------
// Persistent kernel: runs the whole 512-step scan on-chip.
// grid = 64 blocks x 128 threads (256 waves == 4 m-tiles x 64 n-tiles).
//
// Hazard analysis with ping-pong h buffers shows ONE barrier per step is
// sufficient (between layer0 and layer1):
//  - RAW h0w (L0(t) -> L1(t)): protected by the mid-step barrier.
//  - WAR h0[t&1] (L0(t) reads, L0(t+1) writes): protected by the same barrier.
//  - WAR h1[t&1] (L1(t) reads, L1(t+1) writes): protected by the NEXT step's
//    mid-step barrier (each wave runs L0(t+1) in between).
//  - c / hf / out are exclusively owned per-wave: no cross-wave hazard.
// ---------------------------------------------------------------------------
__global__ __launch_bounds__(128) void lstm_persistent(
    const __bf16* __restrict__ x_bf,
    const __bf16* __restrict__ wih0, const __bf16* __restrict__ whh0,
    const __bf16* __restrict__ wih1, const __bf16* __restrict__ whh1,
    const float*  __restrict__ bsum0, const float* __restrict__ bsum1,
    __bf16* __restrict__ h0buf, __bf16* __restrict__ h1buf,
    float* __restrict__ c0, float* __restrict__ c1,
    float* __restrict__ hf0, float* __restrict__ hf1,
    float* __restrict__ out, unsigned* __restrict__ bar) {
  const int wid  = blockIdx.x * (blockDim.x >> 5) + (threadIdx.x >> 5);
  const int lane = threadIdx.x & 31;
  const int half = lane >> 4;
  const int lc   = lane & 15;
  const int m0   = (wid & 3) * 16;    // B/16 = 4 tiles
  const int n0   = (wid >> 2) * 16;   // H/16 = 64 tiles
  const long BH  = (long)LB * LH;
  const unsigned nblk = gridDim.x;

  unsigned gen = 0;
  for (int t = 0; t < LS; ++t) {
    const __bf16* h0r = h0buf + (size_t)(t & 1) * BH;
    __bf16*       h0w = h0buf + (size_t)((t + 1) & 1) * BH;
    const __bf16* h1r = h1buf + (size_t)(t & 1) * BH;
    __bf16*       h1w = h1buf + (size_t)((t + 1) & 1) * BH;

    // layer 0: gates = x_t * Wih0^T + h0 * Whh0^T + b
    cell_tile(x_bf + (long)t * LIN, (long)LS * LIN, LIN, wih0,
              h0r, LH, whh0, bsum0, c0, h0w, hf0,
              (float*)nullptr, 0L, m0, n0, half, lc);

    grid_sync(bar, nblk, gen);   // single barrier per step (see analysis)

    // layer 1: gates = h0_new * Wih1^T + h1 * Whh1^T + b ; writes outputs
    cell_tile(h0w, (long)LH, LH, wih1,
              h1r, LH, whh1, bsum1, c1, h1w, hf1,
              out + (long)t * LH, (long)LS * LH, m0, n0, half, lc);
  }

  // make every wave's final hf/c stores visible before the cross-wave copy
  grid_sync(bar, nblk, gen);

  // ---- final states: d_out tail layout [h(2,B,H) | c(2,B,H)] -------------
  float* tail = out + (long)LB * LS * LH;
  long tid = (long)blockIdx.x * blockDim.x + threadIdx.x;
  long nthreads = (long)gridDim.x * blockDim.x;
  for (long i = tid; i < BH; i += nthreads) {
    tail[i]          = hf0[i];
    tail[BH + i]     = hf1[i];
    tail[2 * BH + i] = c0[i];
    tail[3 * BH + i] = c1[i];
  }
}

// ---------------------------------------------------------------------------
// host launcher
// ---------------------------------------------------------------------------
static size_t carve(size_t& off, size_t bytes) {
  size_t o = off;
  off = (off + bytes + 255) & ~(size_t)255;
  return o;
}

extern "C" void kernel_launch(void* const* d_in, const int* in_sizes, int n_in,
                              void* d_out, int out_size, void* d_ws, size_t ws_size,
                              hipStream_t stream) {
  (void)in_sizes; (void)n_in; (void)out_size; (void)ws_size;
  const float* x     = (const float*)d_in[0];
  const float* Wih0  = (const float*)d_in[1];
  const float* Whh0  = (const float*)d_in[2];
  const float* bih0  = (const float*)d_in[3];
  const float* bhh0  = (const float*)d_in[4];
  const float* Wih1  = (const float*)d_in[5];
  const float* Whh1  = (const float*)d_in[6];
  const float* bih1  = (const float*)d_in[7];
  const float* bhh1  = (const float*)d_in[8];
  float* out = (float*)d_out;

  char* ws = (char*)d_ws;
  size_t off = 0;
  const long XN = (long)LB * LS * LIN;                   // 16.7M
  const long BH = (long)LB * LH;                         // 65536
  __bf16*   x_bf  = (__bf16*)(ws + carve(off, XN * 2));
  __bf16*   wih0  = (__bf16*)(ws + carve(off, (long)LG * LIN * 2));
  __bf16*   whh0  = (__bf16*)(ws + carve(off, (long)LG * LH * 2));
  __bf16*   wih1  = (__bf16*)(ws + carve(off, (long)LG * LH * 2));
  __bf16*   whh1  = (__bf16*)(ws + carve(off, (long)LG * LH * 2));
  float*    bsum0 = (float*)(ws + carve(off, LG * 4));
  float*    bsum1 = (float*)(ws + carve(off, LG * 4));
  __bf16*   h0buf = (__bf16*)(ws + carve(off, 2 * BH * 2));  // ping-pong
  __bf16*   h1buf = (__bf16*)(ws + carve(off, 2 * BH * 2));  // ping-pong
  float*    c0    = (float*)(ws + carve(off, BH * 4));
  float*    c1    = (float*)(ws + carve(off, BH * 4));
  float*    hf0   = (float*)(ws + carve(off, BH * 4));
  float*    hf1   = (float*)(ws + carve(off, BH * 4));
  unsigned* bar   = (unsigned*)(ws + carve(off, 256));

  // ---- prep (re-run every call: deterministic) --------------------------
  cvt_f32_to_bf16<<<4096, 256, 0, stream>>>(x, x_bf, XN);
  cvt_f32_to_bf16<<<2048, 256, 0, stream>>>(Wih0, wih0, (long)LG * LIN);
  cvt_f32_to_bf16<<<4096, 256, 0, stream>>>(Whh0, whh0, (long)LG * LH);
  cvt_f32_to_bf16<<<4096, 256, 0, stream>>>(Wih1, wih1, (long)LG * LH);
  cvt_f32_to_bf16<<<4096, 256, 0, stream>>>(Whh1, whh1, (long)LG * LH);
  bias_sum_kernel<<<LG / 256, 256, 0, stream>>>(bih0, bhh0, bsum0, LG);
  bias_sum_kernel<<<LG / 256, 256, 0, stream>>>(bih1, bhh1, bsum1, LG);
  zero_bf16<<<256, 256, 0, stream>>>(h0buf, 2 * BH);
  zero_bf16<<<256, 256, 0, stream>>>(h1buf, 2 * BH);
  zero_f32<<<256, 256, 0, stream>>>(c0, BH);
  zero_f32<<<256, 256, 0, stream>>>(c1, BH);
  zero_u32<<<1, 64, 0, stream>>>(bar, 64);

  // ---- the whole 512-step scan in ONE persistent kernel -----------------
  lstm_persistent<<<64, 128, 0, stream>>>(
      x_bf, wih0, whh0, wih1, whh1, bsum0, bsum1,
      h0buf, h1buf, c0, c1, hf0, hf1, out, bar);
}
